// SpatiallyAwareEmbedding_50311246905350
// MI455X (gfx1250) — matrix operand order repaired and checked by reading, not compiled, not run on hardware
//
#include <hip/hip_runtime.h>
#include <hip/hip_bf16.h>
#include <math.h>

typedef __attribute__((ext_vector_type(16))) _Float16 v16h;
typedef __attribute__((ext_vector_type(8)))  float    v8f;

#define EPSF 1e-5f

// Branchless exact-GELU: 0.5x(1+erf(x/sqrt(2))) with Abramowitz-Stegun 7.1.26
// erf approximation (max abs err 1.5e-7, same class as erff). Uses the fast
// v_rcp_f32 / v_exp_f32 transcendental units (co-execute with VALU/WMMA) and
// copysign (v_bfi_b32) instead of compare+select for the sign.
__device__ __forceinline__ float gelu_exact(float x) {
    float u = __builtin_fabsf(x) * 0.70710678118654752440f;   // |x|/sqrt(2)
    float t = __builtin_amdgcn_rcpf(__builtin_fmaf(0.3275911f, u, 1.0f));
    float p = __builtin_fmaf(t, 1.061405429f, -1.453152027f);
    p = __builtin_fmaf(t, p, 1.421413741f);
    p = __builtin_fmaf(t, p, -0.284496736f);
    p = __builtin_fmaf(t, p, 0.254829592f);
    p *= t;
    float e = __builtin_amdgcn_exp2f(u * u * -1.4426950408889634f); // exp(-u^2)
    float erfu = __builtin_fmaf(-p, e, 1.0f);                       // erf(|u|) >= 0
    float s = __builtin_copysignf(erfu, x);                         // v_bfi_b32
    return 0.5f * x * (1.0f + s);
}

// One small block: Linear(NIN->8) + bias, exact GELU, LayerNorm(8).
template <int NIN>
__device__ __forceinline__ void block8(const float* in, const float* w, const float* b,
                                       const float* g, const float* be, float* out) {
    float h[8];
#pragma unroll
    for (int o = 0; o < 8; ++o) {
        float a = b[o];
#pragma unroll
        for (int i = 0; i < NIN; ++i) a += w[o * NIN + i] * in[i];
        h[o] = gelu_exact(a);
    }
    float mu = 0.0f;
#pragma unroll
    for (int o = 0; o < 8; ++o) mu += h[o];
    mu *= 0.125f;
    float var = 0.0f;
#pragma unroll
    for (int o = 0; o < 8; ++o) { float d = h[o] - mu; var += d * d; }
    var *= 0.125f;
    float rs = rsqrtf(var + EPSF);
#pragma unroll
    for (int o = 0; o < 8; ++o) out[o] = (h[o] - mu) * rs * g[o] + be[o];
}

// ---------------------------------------------------------------------------
// Prep kernel: f32 -> f16 weight conversion + padding + LN folding.
//   w1pad[384*32]: row n = { f16(f_w1[n][0..15]), f16(f_b1[n]) at K=16, zeros }
//                  (bias rides the WMMA via A's K=16 slot == 1.0)
//   w2h[256*384] = f16(f_w2[e][n] * f_g1[n])       (gamma folded into W2)
//   cb[e] = f_b2[e] + sum_n f_be1[n] * f_w2[e][n]  (beta term + final bias)
//   c2[e] =          sum_n f_g1[n]  * f_w2[e][n]   (mean-correction vector)
// ---------------------------------------------------------------------------
__global__ void prep_kernel(const float* __restrict__ fw1, const float* __restrict__ fb1,
                            const float* __restrict__ fg1, const float* __restrict__ fbe1,
                            const float* __restrict__ fw2, const float* __restrict__ fb2,
                            _Float16* __restrict__ w1pad, _Float16* __restrict__ w2h,
                            float* __restrict__ cb, float* __restrict__ c2) {
    int tid = blockIdx.x * blockDim.x + threadIdx.x;
    if (tid < 12288) {
        int n = tid >> 5, k = tid & 31;
        float v = (k < 16) ? fw1[n * 16 + k] : ((k == 16) ? fb1[n] : 0.0f);
        w1pad[tid] = (_Float16)v;
    }
    if (tid < 98304) {
        int n = tid % 384;
        w2h[tid] = (_Float16)(fw2[tid] * fg1[n]);
    }
    if (tid < 256) {
        float s1 = 0.0f, s2 = 0.0f;
        for (int n = 0; n < 384; ++n) {
            float w = fw2[tid * 384 + n];
            s1 += fbe1[n] * w;
            s2 += fg1[n] * w;
        }
        cb[tid] = s1 + fb2[tid];
        c2[tid] = s2;
    }
}

// ---------------------------------------------------------------------------
// Main fused kernel. 128 threads = 4 waves; 16 tokens per wave; 4096 blocks.
// ---------------------------------------------------------------------------
__global__ __launch_bounds__(128) void fused_embed_kernel(
    const float* __restrict__ nir,
    const float* __restrict__ m_w1, const float* __restrict__ m_b1,
    const float* __restrict__ m_g1, const float* __restrict__ m_be1,
    const float* __restrict__ m_w2, const float* __restrict__ m_b2,
    const float* __restrict__ m_g2, const float* __restrict__ m_be2,
    const float* __restrict__ p_w1, const float* __restrict__ p_b1,
    const float* __restrict__ p_g1, const float* __restrict__ p_be1,
    const float* __restrict__ p_w2, const float* __restrict__ p_b2,
    const float* __restrict__ p_g2, const float* __restrict__ p_be2,
    const _Float16* __restrict__ w1pad, const _Float16* __restrict__ w2h,
    const float* __restrict__ cb, const float* __restrict__ c2,
    float* __restrict__ out) {

    __shared__ __attribute__((aligned(16))) _Float16 cat_s[4 * 16 * 16];   //  2 KB
    __shared__ __attribute__((aligned(16))) _Float16 h_s[4 * 16 * 384];    // 48 KB

    const int wave = threadIdx.x >> 5;
    const int lane = threadIdx.x & 31;
    const int m    = lane & 15;     // row / column-in-tile index
    const int hi   = lane >> 4;     // half-wave index
    const int tokenBase = blockIdx.x * 64 + wave * 16;

    union AB { v16h h; uint4 q[2]; unsigned w[8]; };

    // ---------------- Phase A: small MLP branches (lanes 0..15) ----------------
    if (lane < 16) {
        int token = tokenBase + lane;
        float4 xa = ((const float4*)nir)[token * 2 + 0];
        float4 xb = ((const float4*)nir)[token * 2 + 1];
        float x[8] = {xa.x, xa.y, xa.z, xa.w, xb.x, xb.y, xb.z, xb.w};

        float t1[8], t2[8], q1[8], q2[8];
        block8<2>(x,      m_w1, m_b1, m_g1, m_be1, t1);
        block8<8>(t1,     m_w2, m_b2, m_g2, m_be2, t2);
        block8<6>(x + 2,  p_w1, p_b1, p_g1, p_be1, q1);
        block8<8>(q1,     p_w2, p_b2, p_g2, p_be2, q2);

        _Float16* cs = &cat_s[(wave * 16 + lane) * 16];
#pragma unroll
        for (int i = 0; i < 8; ++i) { cs[i] = (_Float16)t2[i]; cs[8 + i] = (_Float16)q2[i]; }
    }
    __syncthreads();

    // ---------------- Phase B: GEMM1 (cat[16x16+1] x W1pad[32x384]) ------------
    // A operand (16x32 f16): lanes 0-15: v0..3 = K0..7, v4 = {1.0, 0} (bias hook
    // at K=16); lanes 16-31: v0..3 = K8..15, v4..7 = 0.
    AB A;
    {
        const uint4* cp = (const uint4*)&cat_s[(wave * 16 + m) * 16 + hi * 8];
        A.q[0] = cp[0];
        A.w[4] = hi ? 0u : 0x00003C00u;   // f16 1.0 in K=16 slot (lanes 0-15 only)
        A.w[5] = A.w[6] = A.w[7] = 0u;
    }
    v8f zc;
#pragma unroll
    for (int r = 0; r < 8; ++r) zc[r] = 0.0f;

    float sum[8], sq[8];
#pragma unroll
    for (int r = 0; r < 8; ++r) { sum[r] = 0.0f; sq[r] = 0.0f; }

#pragma unroll 4
    for (int t = 0; t < 24; ++t) {
        int n = t * 16 + m;
        AB Bt;   // unconditional: lanes 0-15 get K0..7,16..23, lanes 16-31 get K8..15,24..31
        const uint4* bp = (const uint4*)(w1pad + n * 32 + hi * 8);
        Bt.q[0] = bp[0];
        Bt.q[1] = bp[2];
        v8f acc = __builtin_amdgcn_wmma_f32_16x16x32_f16(false, A.h, false, Bt.h,
                                                         (short)0, zc, false, false);
#pragma unroll
        for (int r = 0; r < 8; ++r) {
            float g = gelu_exact(acc[r]);      // bias already included via K=16 row
            sum[r] += g;
            sq[r]  += g * g;
            h_s[(wave * 16 + r + 8 * hi) * 384 + n] = (_Float16)g;  // row = r+8*hi
        }
    }

    // LN stats per token row: reduce across the 16 lanes of each half-wave.
    float mrs[8], rstd[8];
#pragma unroll
    for (int r = 0; r < 8; ++r) {
        float s = sum[r], s2 = sq[r];
#pragma unroll
        for (int off = 1; off < 16; off <<= 1) {
            s  += __shfl_xor(s,  off, 16);
            s2 += __shfl_xor(s2, off, 16);
        }
        float mu  = s * (1.0f / 384.0f);
        float var = s2 * (1.0f / 384.0f) - mu * mu;
        float rs  = rsqrtf(var + EPSF);
        rstd[r] = rs;
        mrs[r]  = mu * rs;
    }
    __syncthreads();

    // ---------------- Phase C: GEMM2 (g[16x384] x W2'[384x256]) ----------------
    // LN folded: out = rstd*(g @ W2'^T) + cb - (mean*rstd)*c2
#pragma unroll 1
    for (int half = 0; half < 2; ++half) {
        v8f acc[8];
#pragma unroll
        for (int j = 0; j < 8; ++j) {
#pragma unroll
            for (int r = 0; r < 8; ++r) acc[j][r] = 0.0f;
        }
#pragma unroll 2
        for (int kk = 0; kk < 12; ++kk) {
            // A tile: row m, K = kk*32 + (hi? 8..15,24..31 : 0..7,16..23)
            AB Ah;
            const uint4* hp0 = (const uint4*)&h_s[(wave * 16 + m) * 384 + kk * 32 + hi * 8];
            const uint4* hp1 = (const uint4*)&h_s[(wave * 16 + m) * 384 + kk * 32 + 16 + hi * 8];
            Ah.q[0] = hp0[0];
            Ah.q[1] = hp1[0];
#pragma unroll
            for (int j = 0; j < 8; ++j) {
                int n = (half * 8 + j) * 16 + m;   // output channel column
                AB Bh;
                const uint4* wp = (const uint4*)(w2h + n * 384 + kk * 32 + hi * 8);
                Bh.q[0] = wp[0];
                Bh.q[1] = wp[2];
                acc[j] = __builtin_amdgcn_wmma_f32_16x16x32_f16(false, Ah.h, false, Bh.h,
                                                                (short)0, acc[j], false, false);
            }
        }
#pragma unroll
        for (int j = 0; j < 8; ++j) {
            int e = (half * 8 + j) * 16 + m;
            float cbv = cb[e], c2v = c2[e];
#pragma unroll
            for (int r = 0; r < 8; ++r) {
                int token = tokenBase + r + 8 * hi;
                float base = __builtin_fmaf(-mrs[r], c2v, cbv);
                out[token * 256 + e] = __builtin_fmaf(rstd[r], acc[j][r], base);
            }
        }
    }
}

extern "C" void kernel_launch(void* const* d_in, const int* in_sizes, int n_in,
                              void* d_out, int out_size, void* d_ws, size_t ws_size,
                              hipStream_t stream) {
    const float* nir   = (const float*)d_in[0];
    const float* m_w1  = (const float*)d_in[1];
    const float* m_b1  = (const float*)d_in[2];
    const float* m_g1  = (const float*)d_in[3];
    const float* m_be1 = (const float*)d_in[4];
    const float* m_w2  = (const float*)d_in[5];
    const float* m_b2  = (const float*)d_in[6];
    const float* m_g2  = (const float*)d_in[7];
    const float* m_be2 = (const float*)d_in[8];
    const float* p_w1  = (const float*)d_in[9];
    const float* p_b1  = (const float*)d_in[10];
    const float* p_g1  = (const float*)d_in[11];
    const float* p_be1 = (const float*)d_in[12];
    const float* p_w2  = (const float*)d_in[13];
    const float* p_b2  = (const float*)d_in[14];
    const float* p_g2  = (const float*)d_in[15];
    const float* p_be2 = (const float*)d_in[16];
    const float* f_w1  = (const float*)d_in[17];
    const float* f_b1  = (const float*)d_in[18];
    const float* f_g1  = (const float*)d_in[19];
    const float* f_be1 = (const float*)d_in[20];
    const float* f_w2  = (const float*)d_in[21];
    const float* f_b2  = (const float*)d_in[22];

    char* ws = (char*)d_ws;
    _Float16* w1pad = (_Float16*)(ws + 0);       //  24,576 B
    _Float16* w2h   = (_Float16*)(ws + 24576);   // 196,608 B -> 221,184
    float*    cbv   = (float*)(ws + 221184);     //   1,024 B -> 222,208
    float*    c2v   = (float*)(ws + 222208);     //   1,024 B -> 223,232 total

    prep_kernel<<<384, 256, 0, stream>>>(f_w1, f_b1, f_g1, f_be1, f_w2, f_b2,
                                         w1pad, w2h, cbv, c2v);

    // 64*4096 = 262,144 tokens; 64 tokens per block.
    fused_embed_kernel<<<4096, 128, 0, stream>>>(
        nir,
        m_w1, m_b1, m_g1, m_be1, m_w2, m_b2, m_g2, m_be2,
        p_w1, p_b1, p_g1, p_be1, p_w2, p_b2, p_g2, p_be2,
        w1pad, w2h, cbv, c2v, (float*)d_out);
}
